// CircularRelativeAttention_85521388798342
// MI455X (gfx1250) — matrix operand order repaired and checked by reading, not compiled, not run on hardware
//
#include <hip/hip_runtime.h>
#include <hip/hip_bf16.h>

// ---------------------------------------------------------------------------
// Circular relative attention for MI455X (gfx1250, wave32, WMMA).
// All matmuls in bf16 WMMA (16x16x32) with f32 accumulation; softmax f32.
// Attention K/V blocks are staged in double-buffered LDS with
// global_load_async_to_lds_b128 (ASYNCcnt) so the copy of key-block i+1
// overlaps the WMMA/softmax compute of key-block i, and all 8 waves of a
// block share one copy (8x traffic reduction vs per-wave loads).
// The GEMM K-loop is a copy-free 2-stage software pipeline (unroll-by-2,
// alternating fragment register sets) so VMEM latency hides under WMMAs
// without v_mov register rotation.
//
// Workspace layout (48 MB needed):
//   [0,8M)   xb  : x cast to bf16, row-major [4096,1024]   (reused as AO)
//   [8,10M)  wqT : wq^T bf16 [N=1024][K=1024]
//   [10,12M) wkT, [12,14M) wvT, [14,16M) woT
//   [16,24M) qb  : Q bf16 row-major [4096,1024]
//   [24,32M) kbuf: K bf16 row-major
//   [32,40M) vb  : V bf16 row-major (temp)
//   [40,48M) vt  : V transposed per head bf16 [B][H][64][2048]
// ---------------------------------------------------------------------------

#define DEVI __device__ __forceinline__

typedef __attribute__((ext_vector_type(16))) __bf16 v16bf;
typedef __attribute__((ext_vector_type(8)))  float  v8f;

static constexpr int BATCH = 2;
static constexpr int SEQ   = 2048;
static constexpr int DM    = 1024;
static constexpr int NH    = 16;
static constexpr int HD    = 64;
static constexpr int PL    = 4096;   // PERIOD_LEN
static constexpr int BS    = BATCH * SEQ;  // 4096

union FragU { v16bf v; uint4 q[2]; };

DEVI unsigned short f2bf(float f) {
  unsigned u = __float_as_uint(f);
  unsigned r = u + 0x7FFFu + ((u >> 16) & 1u);   // round-to-nearest-even
  return (unsigned short)(r >> 16);
}

// A-matrix fragment (16x32 tile). Per ISA 7.12.2: lanes 0-15 hold K chunks
// {0..7, 16..23}; lanes 16-31 hold {8..15, 24..31}.
DEVI v16bf load_a_frag(const unsigned short* base, int half) {
  const int c0 = half ? 8 : 0;
  FragU f;
  f.q[0] = *reinterpret_cast<const uint4*>(base + c0);
  f.q[1] = *reinterpret_cast<const uint4*>(base + c0 + 16);
  return f.v;
}

// B-matrix fragment (32x16). Lanes 0-15 hold K {0..7, 16..23}, lanes 16-31
// hold K {8..15, 24..31} for column N = lane%16 (column-contiguous source).
DEVI v16bf load_b_frag(const unsigned short* base, int half) {
  const int c0 = half ? 8 : 0;
  FragU f;
  f.q[0] = *reinterpret_cast<const uint4*>(base + c0);
  f.q[1] = *reinterpret_cast<const uint4*>(base + c0 + 16);
  return f.v;
}

DEVI v8f wmma_bf16(v16bf a, v16bf b, v8f c) {
  return __builtin_amdgcn_wmma_f32_16x16x32_bf16(
      /*neg_a=*/false, a, /*neg_b=*/false, b,
      /*c_mod=*/(short)0, c, /*reuse_a=*/false, /*reuse_b=*/false);
}

// Async global->LDS copy of 16 bytes per active lane (GV mode, ASYNCcnt).
DEVI void async_copy_b128(const unsigned short* g, unsigned short* l) {
  const unsigned int  ldsAddr = (unsigned int)(unsigned long long)(uintptr_t)l;
  const unsigned long long ga = (unsigned long long)(uintptr_t)g;
  asm volatile("global_load_async_to_lds_b128 %0, %1, off"
               :: "v"(ldsAddr), "v"(ga) : "memory");
}

DEVI void wait_async_all() {
  asm volatile("s_wait_asynccnt 0x0" ::: "memory");
}

// ---------------------------------------------------------------------------
__global__ void cast_f32_to_bf16_kernel(const float* __restrict__ in,
                                        unsigned short* __restrict__ out, int n) {
  int i = blockIdx.x * blockDim.x + threadIdx.x;
  if (i < n) out[i] = f2bf(in[i]);
}

// wT[n*DM + k] = bf16(w[k*DM + n]) via 32x32 LDS tiles.
__global__ __launch_bounds__(256)
void transpose_cast_kernel(const float* __restrict__ w,
                           unsigned short* __restrict__ wT) {
  __shared__ float tile[32][33];
  const int x = threadIdx.x, y0 = threadIdx.y;
  const int nB = blockIdx.x * 32, kB = blockIdx.y * 32;
  for (int y = y0; y < 32; y += 8)
    tile[y][x] = w[(size_t)(kB + y) * DM + nB + x];
  __syncthreads();
  for (int y = y0; y < 32; y += 8)
    wT[(size_t)(nB + y) * DM + kB + x] = f2bf(tile[x][y]);
}

// vt[((b*NH+h)*HD + d)*SEQ + s] = vb[(b*SEQ+s)*DM + h*HD + d]
__global__ void transpose_v_kernel(const unsigned short* __restrict__ vb,
                                   unsigned short* __restrict__ vt) {
  int idx = blockIdx.x * blockDim.x + threadIdx.x;  // exactly B*H*HD*SEQ threads
  int s = idx & (SEQ - 1);
  int d = (idx >> 11) & (HD - 1);
  int h = (idx >> 17) & (NH - 1);
  int b = idx >> 21;
  vt[idx] = vb[(size_t)(b * SEQ + s) * DM + h * HD + d];
}

// ---------------------------------------------------------------------------
// C[m,n] = sum_k A[m,k] * Bt[n,k] + bias[n].   8 waves: 2(M) x 4(N),
// each wave does a 16x64 tile. Copy-free 2-stage pipeline (requires K%64==0):
// stage-1 loads overlap stage-0 WMMAs and vice versa, no register rotation.
__global__ __launch_bounds__(256)
void gemm_bf16_kernel(const unsigned short* __restrict__ A,
                      const unsigned short* __restrict__ Bt,
                      const float* __restrict__ bias,
                      void* __restrict__ out,
                      int M, int N, int K, int out_f32) {
  const int tid  = threadIdx.x;
  const int wave = tid >> 5, lane = tid & 31;
  const int half = lane >> 4, lm = lane & 15;
  const int mBase = blockIdx.y * 32 + (wave >> 2) * 16;
  const int nBase = blockIdx.x * 256 + (wave & 3) * 64;

  const unsigned short* aRow = A + (size_t)(mBase + lm) * K;
  const unsigned short* bRow[4];
#pragma unroll
  for (int j = 0; j < 4; ++j)
    bRow[j] = Bt + (size_t)(nBase + j * 16 + lm) * K;

  const v8f zero = {0.f, 0.f, 0.f, 0.f, 0.f, 0.f, 0.f, 0.f};
  v8f c[4];
#pragma unroll
  for (int j = 0; j < 4; ++j) c[j] = zero;

  // preheader: stage 0 holds k=0
  v16bf a0 = load_a_frag(aRow, half);
  v16bf b0[4];
#pragma unroll
  for (int j = 0; j < 4; ++j) b0[j] = load_b_frag(bRow[j], half);

  for (int kb = 0; kb < K; kb += 64) {
    // load stage 1 (k+32): always in range since K % 64 == 0
    __builtin_prefetch(aRow + kb + 128, 0, 1);
    v16bf a1 = load_a_frag(aRow + kb + 32, half);
    v16bf b1[4];
#pragma unroll
    for (int j = 0; j < 4; ++j) b1[j] = load_b_frag(bRow[j] + kb + 32, half);

    // compute stage 0 (overlaps stage-1 loads)
#pragma unroll
    for (int j = 0; j < 4; ++j) c[j] = wmma_bf16(a0, b0[j], c[j]);

    // reload stage 0 for next iteration (k+64)
    if (kb + 64 < K) {
      a0 = load_a_frag(aRow + kb + 64, half);
#pragma unroll
      for (int j = 0; j < 4; ++j) b0[j] = load_b_frag(bRow[j] + kb + 64, half);
    }

    // compute stage 1 (overlaps stage-0 loads)
#pragma unroll
    for (int j = 0; j < 4; ++j) c[j] = wmma_bf16(a1, b1[j], c[j]);
  }

#pragma unroll
  for (int j = 0; j < 4; ++j) {
    const int n = nBase + j * 16 + lm;
    const float bv = bias[n];
#pragma unroll
    for (int r = 0; r < 8; ++r) {
      const int m = mBase + r + (half ? 8 : 0);    // C-fragment row mapping
      const float val = c[j][r] + bv;
      if (out_f32) ((float*)out)[(size_t)m * N + n] = val;
      else ((unsigned short*)out)[(size_t)m * N + n] = f2bf(val);
    }
  }
}

// ---------------------------------------------------------------------------
// Flash-style attention: one wave owns 16 q-rows; the whole 8-wave block
// streams 32-key blocks through double-buffered LDS tiles filled with
// async global->LDS copies.
__global__ __launch_bounds__(256)
void attn_kernel(const unsigned short* __restrict__ Q,
                 const unsigned short* __restrict__ Kmat,
                 const unsigned short* __restrict__ Vt,
                 const float* __restrict__ rel_bias,
                 unsigned short* __restrict__ AO) {
  __shared__ float biasCol[PL];                                // 16 KB
  __shared__ __align__(16) unsigned short pTile[8][16 * 40];   // per-wave P
  __shared__ __align__(16) unsigned short kTile[2][32 * 64];   // 2 x 4 KB
  __shared__ __align__(16) unsigned short vTile[2][64 * 32];   // 2 x 4 KB

  const int tid = threadIdx.x;
  const int bh = blockIdx.y;
  const int b = bh >> 4, h = bh & 15;

  for (int i = tid; i < PL; i += 256) biasCol[i] = rel_bias[i * NH + h];

  const int wave = tid >> 5, lane = tid & 31;
  const int half = lane >> 4, lm = lane & 15;
  const int qBase = blockIdx.x * 128 + wave * 16;

  // Q A-fragments for d=0..31 and d=32..63 (loaded once, reused 64 iters)
  const unsigned short* qRow = Q + (size_t)(b * SEQ + qBase + lm) * DM + h * HD;
  const v16bf qa0 = load_a_frag(qRow, half);
  const v16bf qa1 = load_a_frag(qRow + 32, half);

  const unsigned short* kBase = Kmat + (size_t)(b * SEQ) * DM + h * HD;
  const unsigned short* vBase = Vt + (size_t)(b * NH + h) * HD * SEQ;

  // async staging assignment: 256 x 16B chunks for K tile, 128 for V tile
  const int kr = tid >> 2, kc = (tid & 3) * 8;   // K: row=key, 4 chunks/row
  const int vd = tid >> 1, vc = (tid & 1) * 8;   // V: row=d,   2 chunks/row

  float mrow[8], lrow[8];
#pragma unroll
  for (int r = 0; r < 8; ++r) { mrow[r] = -1e30f; lrow[r] = 0.f; }
  const v8f zero = {0.f, 0.f, 0.f, 0.f, 0.f, 0.f, 0.f, 0.f};
  v8f o[4];
#pragma unroll
  for (int j = 0; j < 4; ++j) o[j] = zero;

  unsigned short* pW = &pTile[wave][0];

  // prologue: stage key-block 0 into buffer 0
  async_copy_b128(kBase + (size_t)kr * DM + kc, &kTile[0][kr * 64 + kc]);
  if (tid < 128)
    async_copy_b128(vBase + (size_t)vd * SEQ + vc, &vTile[0][vd * 32 + vc]);
  wait_async_all();
  __syncthreads();   // also publishes biasCol

  for (int kb = 0, it = 0; kb < SEQ; kb += 32, ++it) {
    const int cur = it & 1;
    // stage next key-block while computing this one
    if (kb + 32 < SEQ) {
      const int nxt = cur ^ 1;
      async_copy_b128(kBase + (size_t)(kb + 32 + kr) * DM + kc,
                      &kTile[nxt][kr * 64 + kc]);
      if (tid < 128)
        async_copy_b128(vBase + (size_t)vd * SEQ + (kb + 32) + vc,
                        &vTile[nxt][vd * 32 + vc]);
    }

    const unsigned short* kT = &kTile[cur][0];
    const unsigned short* vT = &vTile[cur][0];

    // scores: S = Q @ K^T  (B-fragment column n = key row, LDS-contiguous)
    v8f s0 = zero, s1 = zero;
    s0 = wmma_bf16(qa0, load_b_frag(kT + lm * 64, half), s0);
    s0 = wmma_bf16(qa1, load_b_frag(kT + lm * 64 + 32, half), s0);
    s1 = wmma_bf16(qa0, load_b_frag(kT + (16 + lm) * 64, half), s1);
    s1 = wmma_bf16(qa1, load_b_frag(kT + (16 + lm) * 64 + 32, half), s1);

    // scale + circular bias + online softmax (rows live in 16-lane halves)
#pragma unroll
    for (int r = 0; r < 8; ++r) {
      const int q = qBase + r + (half ? 8 : 0);
      const int k0 = kb + lm, k1 = k0 + 16;
      float v0 = s0[r] * 0.125f + biasCol[(q - k0) & (PL - 1)];
      float v1 = s1[r] * 0.125f + biasCol[(q - k1) & (PL - 1)];
      float mx = fmaxf(v0, v1);
      mx = fmaxf(mx, __shfl_xor(mx, 1, 32));
      mx = fmaxf(mx, __shfl_xor(mx, 2, 32));
      mx = fmaxf(mx, __shfl_xor(mx, 4, 32));
      mx = fmaxf(mx, __shfl_xor(mx, 8, 32));
      const float mnew  = fmaxf(mrow[r], mx);
      const float alpha = __expf(mrow[r] - mnew);
      const float p0 = __expf(v0 - mnew);
      const float p1 = __expf(v1 - mnew);
      float ps = p0 + p1;
      ps += __shfl_xor(ps, 1, 32);
      ps += __shfl_xor(ps, 2, 32);
      ps += __shfl_xor(ps, 4, 32);
      ps += __shfl_xor(ps, 8, 32);
      lrow[r] = lrow[r] * alpha + ps;
      mrow[r] = mnew;
#pragma unroll
      for (int j = 0; j < 4; ++j) o[j][r] *= alpha;
      // stage P (C-fragment layout -> LDS) for re-load as an A-fragment
      const int prow = r + (half ? 8 : 0);
      pW[prow * 40 + lm]      = f2bf(p0);
      pW[prow * 40 + 16 + lm] = f2bf(p1);
    }
    asm volatile("s_wait_dscnt 0x0" ::: "memory");  // cross-lane LDS RAW, same wave
    const v16bf pf = load_a_frag(pW + lm * 40, half);
    asm volatile("" ::: "memory");                  // keep next stores after loads

    // O += P @ V  (Vt columns d = j*16 + lane%16, LDS-contiguous over keys)
#pragma unroll
    for (int j = 0; j < 4; ++j)
      o[j] = wmma_bf16(pf, load_b_frag(vT + (j * 16 + lm) * 32, half), o[j]);

    // next buffer must be fully staged before anyone reads it
    wait_async_all();
    __syncthreads();
  }

  // normalize and store (bf16, pre-projection activation)
#pragma unroll
  for (int r = 0; r < 8; ++r) {
    const float inv = 1.0f / lrow[r];
    const int q = qBase + r + (half ? 8 : 0);
    const size_t base = (size_t)(b * SEQ + q) * DM + h * HD;
#pragma unroll
    for (int j = 0; j < 4; ++j)
      AO[base + j * 16 + lm] = f2bf(o[j][r] * inv);
  }
}

// ---------------------------------------------------------------------------
extern "C" void kernel_launch(void* const* d_in, const int* in_sizes, int n_in,
                              void* d_out, int out_size, void* d_ws, size_t ws_size,
                              hipStream_t stream) {
  (void)in_sizes; (void)n_in; (void)out_size; (void)ws_size;
  const float* x  = (const float*)d_in[0];
  const float* wq = (const float*)d_in[1];
  const float* bq = (const float*)d_in[2];
  const float* wk = (const float*)d_in[3];
  const float* bk = (const float*)d_in[4];
  const float* wv = (const float*)d_in[5];
  const float* bv = (const float*)d_in[6];
  const float* wo = (const float*)d_in[7];
  const float* bo = (const float*)d_in[8];
  const float* rb = (const float*)d_in[9];

  char* ws = (char*)d_ws;
  const size_t MB = 1024ull * 1024ull;
  unsigned short* xb   = (unsigned short*)(ws + 0 * MB);   // reused as AO
  unsigned short* wqT  = (unsigned short*)(ws + 8 * MB);
  unsigned short* wkT  = (unsigned short*)(ws + 10 * MB);
  unsigned short* wvT  = (unsigned short*)(ws + 12 * MB);
  unsigned short* woT  = (unsigned short*)(ws + 14 * MB);
  unsigned short* qb   = (unsigned short*)(ws + 16 * MB);
  unsigned short* kbuf = (unsigned short*)(ws + 24 * MB);
  unsigned short* vb   = (unsigned short*)(ws + 32 * MB);
  unsigned short* vt   = (unsigned short*)(ws + 40 * MB);
  unsigned short* ao   = xb;

  // 1) precision conversion / weight transposes
  cast_f32_to_bf16_kernel<<<(BS * DM) / 256, 256, 0, stream>>>(x, xb, BS * DM);
  dim3 tb(32, 8), tg(DM / 32, DM / 32);
  transpose_cast_kernel<<<tg, tb, 0, stream>>>(wq, wqT);
  transpose_cast_kernel<<<tg, tb, 0, stream>>>(wk, wkT);
  transpose_cast_kernel<<<tg, tb, 0, stream>>>(wv, wvT);
  transpose_cast_kernel<<<tg, tb, 0, stream>>>(wo, woT);

  // 2) QKV projections (WMMA GEMMs)
  dim3 gg(DM / 256, BS / 32);  // (4, 128)
  gemm_bf16_kernel<<<gg, 256, 0, stream>>>(xb, wqT, bq, qb,   BS, DM, DM, 0);
  gemm_bf16_kernel<<<gg, 256, 0, stream>>>(xb, wkT, bk, kbuf, BS, DM, DM, 0);
  gemm_bf16_kernel<<<gg, 256, 0, stream>>>(xb, wvT, bv, vb,   BS, DM, DM, 0);

  // 3) per-head V transpose for column-contiguous PV B-fragments
  transpose_v_kernel<<<(BATCH * NH * HD * SEQ) / 256, 256, 0, stream>>>(vb, vt);

  // 4) streaming attention with circular relative bias (async-LDS staged)
  attn_kernel<<<dim3(SEQ / 128, BATCH * NH), 256, 0, stream>>>(qb, kbuf, vt, rb, ao);

  // 5) output projection, f32 result
  gemm_bf16_kernel<<<gg, 256, 0, stream>>>(ao, woT, bo, d_out, BS, DM, DM, 1);
}
